// MultiHeadAttention__15710990369546
// MI455X (gfx1250) — matrix operand order repaired and checked by reading, not compiled, not run on hardware
//
#include <hip/hip_runtime.h>
#include <hip/hip_bf16.h>
#include <math.h>
#include <stdint.h>

// ---------------------------------------------------------------------------
// MI455X (gfx1250) multi-head attention:
//   stage 0: fp32 -> f16 conversion of activations, weight transposition
//   stage 1: Q/K/V projection GEMMs via v_wmma_f32_16x16x32_f16
//   stage 2: flash attention; K/V tiles staged into LDS by the Tensor Data
//            Mover (tensor_load_to_lds, s_wait_tensorcnt), double-buffered
//   stage 3: output projection GEMM (fp32 out + bias)
// ---------------------------------------------------------------------------

typedef __attribute__((ext_vector_type(16))) _Float16 v16h;
typedef __attribute__((ext_vector_type(8)))  float    v8f;
typedef __attribute__((ext_vector_type(4)))  _Float16 v4h;
typedef unsigned int u32x4 __attribute__((ext_vector_type(4)));
typedef int          i32x8 __attribute__((ext_vector_type(8)));
typedef int          i32x4 __attribute__((ext_vector_type(4)));

#define HEADS 8
#define DHEAD 64
#define DMODEL 512
#define DOUT 512
#define SEQ 2048
#define BATCH 4
#define ROWS (BATCH * SEQ)          // 8192
#define LOG2E 1.44269504088896340736f

static __device__ __forceinline__ v8f wmma_f16(v16h a, v16h b, v8f c) {
  // D = A(16x32 f16) * B(32x16 f16) + C(16x16 f32)
  return __builtin_amdgcn_wmma_f32_16x16x32_f16(
      /*neg_a=*/false, a, /*neg_b=*/false, b,
      /*c_mod=*/(short)0, c, /*reuse_a=*/false, /*reuse_b=*/false);
}

// A fragment (16x32, row-major source, rowStride in halves).
// Lane l holds row l%16; halves j -> K = kbase + 16*hw + j
// (contiguous k order, matching load_b_frag)
static __device__ __forceinline__ v16h load_a_frag(const _Float16* base, int rowStride,
                                                   int row, int kbase, int hw) {
  const _Float16* p = base + (size_t)row * rowStride + kbase + 16 * hw;
  v16h a;
  uint4* u = reinterpret_cast<uint4*>(&a);
  u[0] = *reinterpret_cast<const uint4*>(p);
  u[1] = *reinterpret_cast<const uint4*>(p + 8);
  return a;
}

// B fragment (32x16) loaded from the TRANSPOSED operand Bt[col][k] row-major.
// Lane l holds col l%16; halves j -> K = kbase + 16*hw + j.
static __device__ __forceinline__ v16h load_b_frag(const _Float16* base, int rowStride,
                                                   int col, int kbase, int hw) {
  const _Float16* p = base + (size_t)col * rowStride + kbase + 16 * hw;
  v16h b;
  uint4* u = reinterpret_cast<uint4*>(&b);
  const uint4* q = reinterpret_cast<const uint4*>(p);
  u[0] = q[0];
  u[1] = q[1];
  return b;
}

// Low 32 bits of a generic pointer into the LDS aperture == LDS byte offset.
static __device__ __forceinline__ unsigned lds_off(const void* p) {
  return (unsigned)(uintptr_t)p;
}

// Tensor Data Mover: async copy of a 2D f16 tile (tile0 x tile1 elements,
// source row stride stride0 elements) from global to LDS.
// D# packing per CDNA5 ISA 8.3/8.4; groups 2/3 zero (2D tensor).
// 6-arg builtin form (clang-23 lane).
static __device__ __forceinline__ void tdm_load_2d(unsigned ldsByteOff,
                                                   const _Float16* gsrc,
                                                   int tile0, int tile1, int stride0) {
  unsigned long long ga = (unsigned long long)(uintptr_t)gsrc;
  u32x4 g0;
  g0[0] = 1u;                                                   // count=1 (valid)
  g0[1] = ldsByteOff;                                           // lds_addr
  g0[2] = (unsigned)ga;                                         // global_addr[31:0]
  g0[3] = (unsigned)((ga >> 32) & 0x01ffffffu) | (2u << 30);    // addr[56:32]|type=2
  i32x8 g1;
  g1[0] = 1 << 16;                       // wg_mask=0, data_size=1 (2-byte elems)
  g1[1] = tile0 << 16;                   // tensor_dim0[15:0]
  g1[2] = (tile1 & 0xffff) << 16;        // tensor_dim0 hi | tensor_dim1 lo
  g1[3] = (tile0 & 0xffff) << 16;        // tensor_dim1 hi | tile_dim0
  g1[4] = (tile1 & 0xffff);              // tile_dim1 | tile_dim2=0
  g1[5] = stride0;                       // tensor_dim0_stride[31:0]
  g1[6] = 0;
  g1[7] = 0;
  i32x4 z4 = {0, 0, 0, 0};
  i32x8 z8 = {0, 0, 0, 0, 0, 0, 0, 0};
  __builtin_amdgcn_tensor_load_to_lds(g0, g1, z4, z4, z8, /*cpol=*/0);
}

// ---------------------------------------------------------------------------
// Stage 0 kernels
// ---------------------------------------------------------------------------

__global__ __launch_bounds__(256) void cvt_f16_kernel(const float* __restrict__ in,
                                                      _Float16* __restrict__ out, int n4) {
  int i = blockIdx.x * 256 + threadIdx.x;
  if (i >= n4) return;
  float4 v = reinterpret_cast<const float4*>(in)[i];
  v4h o;
  o.x = (_Float16)v.x; o.y = (_Float16)v.y;
  o.z = (_Float16)v.z; o.w = (_Float16)v.w;
  reinterpret_cast<v4h*>(out)[i] = o;
}

// qkv kernels: [H, DM, DH] -> Wt[c = h*64+o][i], f16
__global__ __launch_bounds__(256) void prep_wqkv_kernel(const float* __restrict__ k3,
                                                        _Float16* __restrict__ wt) {
  int idx = blockIdx.x * 256 + threadIdx.x;      // 0 .. 512*512-1
  int c = idx >> 9, i = idx & 511;
  int h = c >> 6, o = c & 63;
  wt[idx] = (_Float16)k3[((size_t)h * DMODEL + i) * DHEAD + o];
}

// projection kernel: [H, DH, DO] -> Wt[o][c2 = h*64+i], f16
__global__ __launch_bounds__(256) void prep_wp_kernel(const float* __restrict__ pk,
                                                      _Float16* __restrict__ wt) {
  int idx = blockIdx.x * 256 + threadIdx.x;      // 0 .. 512*512-1
  int o = idx >> 9, c2 = idx & 511;
  int h = c2 >> 6, i = c2 & 63;
  wt[idx] = (_Float16)pk[((size_t)h * DHEAD + i) * DOUT + o];
}

// ---------------------------------------------------------------------------
// Stage 1 / 3: GEMM  C[8192,512] = A[8192,512] x Wt^T  (+bias, layout per mode)
//   mode 0: Q proj  -> f16 [B,H,N,DH], val = (acc+bias)*scale (scale=1/8)
//   mode 1: K proj  -> f16 [B,H,M,DH]
//   mode 2: V proj  -> f16 [B,H,DH,M]  (transposed for PV B-fragments)
//   mode 3: out proj-> f32 [8192,512]
// Block: 128 threads (4 waves); block tile 64x128; wave tile 16x128
// (8 accumulators -> 8 independent WMMAs per k-step hide load latency).
// ---------------------------------------------------------------------------
__global__ __launch_bounds__(128) void gemm_proj_kernel(const _Float16* __restrict__ A,
                                                        const _Float16* __restrict__ Wt,
                                                        const float* __restrict__ bias,
                                                        void* __restrict__ out,
                                                        int mode, float scale) {
  const int lane = threadIdx.x & 31;
  const int wv   = threadIdx.x >> 5;
  const int l16  = lane & 15;
  const int hw   = lane >> 4;
  const int rowBase = blockIdx.y * 64 + wv * 16;
  const int colBase = blockIdx.x * 128;

  const v8f vzero = {0.f, 0.f, 0.f, 0.f, 0.f, 0.f, 0.f, 0.f};
  v8f acc[8];
#pragma unroll
  for (int t = 0; t < 8; ++t) acc[t] = vzero;

  for (int ks = 0; ks < DMODEL / 32; ++ks) {       // 16 k-steps of 32
    const int kb = ks * 32;
    v16h a = load_a_frag(A, DMODEL, rowBase + l16, kb, hw);
#pragma unroll
    for (int t = 0; t < 8; ++t) {
      v16h b = load_b_frag(Wt, DMODEL, colBase + t * 16 + l16, kb, hw);
      acc[t] = wmma_f16(a, b, acc[t]);
    }
  }

#pragma unroll
  for (int r = 0; r < 8; ++r) {
    const int gr = rowBase + r + 8 * hw;           // global row (b*2048+n)
    const int bb = gr >> 11;
    const int n  = gr & 2047;
#pragma unroll
    for (int t = 0; t < 8; ++t) {
      const int c = colBase + t * 16 + l16;        // output column
      const float v = (acc[t][r] + bias[c]) * scale;
      if (mode == 3) {
        reinterpret_cast<float*>(out)[(size_t)gr * DMODEL + c] = v;
      } else {
        const int h = c >> 6, d = c & 63;
        _Float16* O = reinterpret_cast<_Float16*>(out);
        if (mode == 2) {                           // Vt: [bh][d][m]
          O[((size_t)(bb * HEADS + h) * DHEAD + d) * SEQ + n] = (_Float16)v;
        } else {                                   // Q/K: [bh][n][d]
          O[((size_t)(bb * HEADS + h) * SEQ + n) * DHEAD + d] = (_Float16)v;
        }
      }
    }
  }
}

// ---------------------------------------------------------------------------
// Stage 2: flash attention per (b,h).  Grid (N/64, B*H), block 128 (4 waves).
// Each wave owns 16 query rows (Q fragments hoisted out of the key loop).
// K/V 64x64 tiles are DMA'd into double-buffered LDS by the TDM; one wave
// drives the TDM one tile ahead, s_wait_tensorcnt + barrier publishes them.
// ---------------------------------------------------------------------------
__global__ __launch_bounds__(128) void flash_attn_kernel(const _Float16* __restrict__ Qp,
                                                         const _Float16* __restrict__ Kp,
                                                         const _Float16* __restrict__ Vt,
                                                         _Float16* __restrict__ MH) {
  __shared__ _Float16 ldsK[2][64][64];             // 2 x 8 KB key tiles [m][d]
  __shared__ _Float16 ldsV[2][64][64];             // 2 x 8 KB value tiles [d][m]
  __shared__ _Float16 ldsP[4][16][64];             // per-wave prob tile (8 KB)

  const int lane = threadIdx.x & 31;
  const int wv   = threadIdx.x >> 5;
  const int l16  = lane & 15;
  const int hw   = lane >> 4;

  const int bh = blockIdx.y;                       // b*8 + h
  const int b  = bh >> 3;
  const int h  = bh & 7;
  const _Float16* Q = Qp + (size_t)bh * SEQ * DHEAD;
  const _Float16* K = Kp + (size_t)bh * SEQ * DHEAD;
  const _Float16* V = Vt + (size_t)bh * DHEAD * SEQ;
  const int nBase = blockIdx.x * 64 + wv * 16;

  // ---- Q fragments are loop-invariant: load once ----
  v16h aq[2];
  aq[0] = load_a_frag(Q, DHEAD, nBase + l16, 0, hw);
  aq[1] = load_a_frag(Q, DHEAD, nBase + l16, 32, hw);

  const v8f vzero = {0.f, 0.f, 0.f, 0.f, 0.f, 0.f, 0.f, 0.f};
  v8f o[4];
  float mrow[8], lsum[8];
#pragma unroll
  for (int t = 0; t < 4; ++t) o[t] = vzero;
#pragma unroll
  for (int r = 0; r < 8; ++r) { mrow[r] = -INFINITY; lsum[r] = 0.f; }

  // ---- prologue: TDM tile 0 into buffer 0 (wave 0 drives the TDM) ----
  if (wv == 0) {
    tdm_load_2d(lds_off(&ldsK[0][0][0]), K, 64, 64, DHEAD);
    tdm_load_2d(lds_off(&ldsV[0][0][0]), V, 64, 64, SEQ);
  }

  for (int mt = 0; mt < SEQ; mt += 64) {
    const int buf = (mt >> 6) & 1;
    __syncthreads();                               // prev readers done with buf^1
    if (wv == 0) {
      if (mt + 64 < SEQ) {                         // TDM next tile, then wait for
        tdm_load_2d(lds_off(&ldsK[buf ^ 1][0][0]), K + (size_t)(mt + 64) * DHEAD,
                    64, 64, DHEAD);
        tdm_load_2d(lds_off(&ldsV[buf ^ 1][0][0]), V + (mt + 64), 64, 64, SEQ);
        __builtin_amdgcn_s_wait_tensorcnt(2);      // current pair complete
      } else {
        __builtin_amdgcn_s_wait_tensorcnt(0);
      }
    }
    __syncthreads();                               // current buf published

    // ---- S = Q * K^T  (Q pre-scaled by 1/sqrt(DH)); K frags from LDS ----
    v8f s[4];
#pragma unroll
    for (int t = 0; t < 4; ++t) s[t] = vzero;
#pragma unroll
    for (int ks = 0; ks < 2; ++ks) {               // DH = 64 -> 2 k-steps
#pragma unroll
      for (int t = 0; t < 4; ++t) {
        v16h bfrag = load_b_frag(&ldsK[buf][0][0], 64, t * 16 + l16, ks * 32, hw);
        s[t] = wmma_f16(aq[ks], bfrag, s[t]);
      }
    }

    // ---- online softmax (row = r + 8*hw, 16 lanes per row) ----
    float alpha[8];
#pragma unroll
    for (int r = 0; r < 8; ++r) {
      float mx = fmaxf(fmaxf(s[0][r], s[1][r]), fmaxf(s[2][r], s[3][r]));
      mx = fmaxf(mx, __shfl_xor(mx, 1, 32));
      mx = fmaxf(mx, __shfl_xor(mx, 2, 32));
      mx = fmaxf(mx, __shfl_xor(mx, 4, 32));
      mx = fmaxf(mx, __shfl_xor(mx, 8, 32));
      const float mnew = fmaxf(mrow[r], mx);
      alpha[r] = exp2f((mrow[r] - mnew) * LOG2E);
      mrow[r] = mnew;
      float ps = 0.f;
#pragma unroll
      for (int t = 0; t < 4; ++t) {
        float p = exp2f((s[t][r] - mnew) * LOG2E);
        s[t][r] = p;
        ps += p;
      }
      ps += __shfl_xor(ps, 1, 32);
      ps += __shfl_xor(ps, 2, 32);
      ps += __shfl_xor(ps, 4, 32);
      ps += __shfl_xor(ps, 8, 32);
      lsum[r] = lsum[r] * alpha[r] + ps;
#pragma unroll
      for (int t = 0; t < 4; ++t) o[t][r] = o[t][r] * alpha[r];
    }

    // ---- C-layout -> A-layout for P via LDS ----
#pragma unroll
    for (int t = 0; t < 4; ++t)
#pragma unroll
      for (int r = 0; r < 8; ++r)
        ldsP[wv][r + 8 * hw][t * 16 + l16] = (_Float16)s[t][r];
    __syncthreads();

    // ---- O += P * V  (both operands from LDS) ----
#pragma unroll
    for (int ks = 0; ks < 2; ++ks) {               // 64 keys -> 2 k-steps
      v16h a = load_a_frag(&ldsP[wv][0][0], 64, l16, ks * 32, hw);
#pragma unroll
      for (int t = 0; t < 4; ++t) {
        v16h bfrag = load_b_frag(&ldsV[buf][0][0], 64, t * 16 + l16, ks * 32, hw);
        o[t] = wmma_f16(a, bfrag, o[t]);
      }
    }
  }

  // ---- normalize and write MH[gr][h*64 + d] (f16) ----
#pragma unroll
  for (int r = 0; r < 8; ++r) {
    const float inv = 1.f / lsum[r];
    const int gr = b * SEQ + nBase + r + 8 * hw;
#pragma unroll
    for (int t = 0; t < 4; ++t)
      MH[(size_t)gr * DMODEL + h * DHEAD + t * 16 + l16] = (_Float16)(o[t][r] * inv);
  }
}

// ---------------------------------------------------------------------------
// Host-side launcher
// ---------------------------------------------------------------------------
extern "C" void kernel_launch(void* const* d_in, const int* in_sizes, int n_in,
                              void* d_out, int out_size, void* d_ws, size_t ws_size,
                              hipStream_t stream) {
  const float* query = (const float*)d_in[0];
  const float* key   = (const float*)d_in[1];
  const float* value = (const float*)d_in[2];
  const float* qk    = (const float*)d_in[3];
  const float* kk    = (const float*)d_in[4];
  const float* vk    = (const float*)d_in[5];
  const float* pk    = (const float*)d_in[6];
  const float* qb    = (const float*)d_in[7];
  const float* kb    = (const float*)d_in[8];
  const float* vb    = (const float*)d_in[9];
  const float* pb    = (const float*)d_in[10];

  const size_t SZ_X = (size_t)ROWS * DMODEL * sizeof(_Float16);   // 8 MiB
  const size_t SZ_W = (size_t)DMODEL * DMODEL * sizeof(_Float16); // 512 KiB

  char* ws = (char*)d_ws;
  _Float16* Xq  = (_Float16*)(ws + 0 * SZ_X);   // reused as MH after Q-proj
  _Float16* Xk  = (_Float16*)(ws + 1 * SZ_X);
  _Float16* Xv  = (_Float16*)(ws + 2 * SZ_X);
  _Float16* Qp  = (_Float16*)(ws + 3 * SZ_X);
  _Float16* Kp  = (_Float16*)(ws + 4 * SZ_X);
  _Float16* Vtb = (_Float16*)(ws + 5 * SZ_X);
  _Float16* Wtq = (_Float16*)(ws + 6 * SZ_X + 0 * SZ_W);
  _Float16* Wtk = (_Float16*)(ws + 6 * SZ_X + 1 * SZ_W);
  _Float16* Wtv = (_Float16*)(ws + 6 * SZ_X + 2 * SZ_W);
  _Float16* Wtp = (_Float16*)(ws + 6 * SZ_X + 3 * SZ_W);
  _Float16* MH  = Xq;                            // alias: Xq dead after Q GEMM

  const int n4 = ROWS * DMODEL / 4;              // 1,048,576
  const dim3 cvtGrid(n4 / 256);
  cvt_f16_kernel<<<cvtGrid, 256, 0, stream>>>(query, Xq, n4);
  cvt_f16_kernel<<<cvtGrid, 256, 0, stream>>>(key,   Xk, n4);
  cvt_f16_kernel<<<cvtGrid, 256, 0, stream>>>(value, Xv, n4);

  const dim3 wGrid((DMODEL * DMODEL) / 256);     // 1024 blocks
  prep_wqkv_kernel<<<wGrid, 256, 0, stream>>>(qk, Wtq);
  prep_wqkv_kernel<<<wGrid, 256, 0, stream>>>(kk, Wtk);
  prep_wqkv_kernel<<<wGrid, 256, 0, stream>>>(vk, Wtv);
  prep_wp_kernel<<<wGrid, 256, 0, stream>>>(pk, Wtp);

  const dim3 gemmGrid(DMODEL / 128, ROWS / 64);  // (4, 128)
  gemm_proj_kernel<<<gemmGrid, 128, 0, stream>>>(Xq, Wtq, qb, (void*)Qp, 0, 0.125f);
  gemm_proj_kernel<<<gemmGrid, 128, 0, stream>>>(Xk, Wtk, kb, (void*)Kp, 1, 1.0f);
  gemm_proj_kernel<<<gemmGrid, 128, 0, stream>>>(Xv, Wtv, vb, (void*)Vtb, 2, 1.0f);

  const dim3 faGrid(SEQ / 64, BATCH * HEADS);    // (32, 32)
  flash_attn_kernel<<<faGrid, 128, 0, stream>>>(Qp, Kp, Vtb, MH);

  gemm_proj_kernel<<<gemmGrid, 128, 0, stream>>>(MH, Wtp, pb, d_out, 3, 1.0f);
}
